// Loss_yolov3_89343909691646
// MI455X (gfx1250) — compile-verified, matrix operand order checked by reading, not run on hardware
//
#include <hip/hip_runtime.h>
#include <math.h>

// ---------------- constants (match reference) ----------------
constexpr int   Jg   = 60;     // gt boxes per image
constexpr int   KSEL = 15;     // top-k per (b,j,layer)
constexpr int   NC   = 80;     // classes
constexpr int   Bb   = 16;     // batch
constexpr int   CAP  = 96;     // bucket capacity (mean ~34, >16 sigma)
constexpr int   EMIT = 30;     // EM iterations
constexpr float EPSf = 1e-10f;

typedef __attribute__((ext_vector_type(16))) _Float16 v16h;
typedef __attribute__((ext_vector_type(8)))  float    v8f;
union V16H { v16h v; _Float16 h[16]; };
union V8F  { v8f  v; float    f[8];  };

// fast softplus: 2 TRANS (v_exp_f32, v_log_f32) + 3 VALU
__device__ __forceinline__ float softplusf(float x) {
    return fmaxf(x, 0.f) + __logf(1.f + __expf(-fabsf(x)));
}
__device__ __forceinline__ float wsum(float v) {
    #pragma unroll
    for (int o = 16; o > 0; o >>= 1) v += __shfl_xor(v, o, 32);
    return v;
}
__device__ __forceinline__ float wmin(float v) {
    #pragma unroll
    for (int o = 16; o > 0; o >>= 1) v = fminf(v, __shfl_xor(v, o, 32));
    return v;
}
__device__ __forceinline__ float wmax(float v) {
    #pragma unroll
    for (int o = 16; o > 0; o >>= 1) v = fmaxf(v, __shfl_xor(v, o, 32));
    return v;
}

// ============================================================
// K1: per-anchor pass. One wave = 32 contiguous anchors.
// Stage p/label records into LDS via async b128 loads, then:
//   lane-per-anchor: CIoU / focal terms; WMMA(ones-B) row-sums the
//   80-class BCE for 16 anchors per tile; the two tiles are
//   interleaved so VALU conversion of one tile's A fragment
//   co-executes with the other tile's XDL WMMA.
// Outputs 5 floats per anchor into L2-resident workspace.
// ============================================================
__global__ __launch_bounds__(64) void k_anchor(
    const float* __restrict__ p, const float4* __restrict__ pd,
    const float* __restrict__ label,
    float* __restrict__ o_loss, float* __restrict__ o_aiou,
    float* __restrict__ o_cls,  float* __restrict__ o_f0,
    float* __restrict__ o_f1,   int total, float img2)
{
    __shared__ __align__(16) float sP[2][85 * 32];
    __shared__ __align__(16) float sL[2][86 * 32];
    __shared__ float sX[2][32];

    const int w    = threadIdx.x >> 5;
    const int lane = threadIdx.x & 31;
    const long long wave = (long long)blockIdx.x * 2 + w;
    const long long n0   = wave * 32;
    if (n0 >= total) return;                       // wave-uniform
    const int nstage = min(32, (int)(total - n0)); // multiple of 16

    // ---- async stage: coalesced 128b global -> LDS ----
    {
        const float* src = p + n0 * 85;
        const int nq = (nstage * 85) >> 2;         // float4 count (exact)
        for (int i = lane; i < nq; i += 32) {
            unsigned lo = (unsigned)(unsigned long long)(&sP[w][i * 4]);
            const float* g = src + i * 4;
            asm volatile("global_load_async_to_lds_b128 %0, %1, off"
                         :: "v"(lo), "v"(g) : "memory");
        }
        const float* src2 = label + n0 * 86;
        const int nq2 = (nstage * 86) >> 2;
        for (int i = lane; i < nq2; i += 32) {
            unsigned lo = (unsigned)(unsigned long long)(&sL[w][i * 4]);
            const float* g = src2 + i * 4;
            asm volatile("global_load_async_to_lds_b128 %0, %1, off"
                         :: "v"(lo), "v"(g) : "memory");
        }
        asm volatile("s_wait_asynccnt 0" ::: "memory");
    }

    // ---- lane-per-anchor scalar terms ----
    const int a = min(lane, nstage - 1);           // clamped tail (stores guarded)
    float4 pdv = pd[n0 + a];                       // coalesced 16B/lane
    const float* Lr = &sL[w][a * 86];
    const float lx = Lr[0], ly = Lr[1], lw2 = Lr[2], lh2 = Lr[3], mix = Lr[5];
    const float px = pdv.x, py = pdv.y, pw = pdv.z, ph = pdv.w;

    float p_x1 = px - pw * .5f, p_x2 = px + pw * .5f;
    float p_y1 = py - ph * .5f, p_y2 = py + ph * .5f;
    float l_x1 = lx - lw2 * .5f, l_x2 = lx + lw2 * .5f;
    float l_y1 = ly - lh2 * .5f, l_y2 = ly + lh2 * .5f;
    float iw = fmaxf(fminf(p_x2, l_x2) - fmaxf(p_x1, l_x1), 0.f);
    float ih = fmaxf(fminf(p_y2, l_y2) - fmaxf(p_y1, l_y1), 0.f);
    float inter = iw * ih;
    float iou = inter / (pw * ph + lw2 * lh2 - inter + EPSf);
    float cw = fmaxf(p_x2, l_x2) - fminf(p_x1, l_x1);
    float chh = fmaxf(p_y2, l_y2) - fminf(p_y1, l_y1);
    float c2 = cw * cw + chh * chh + EPSf;
    float rho2 = (px - lx) * (px - lx) + (py - ly) * (py - ly);
    float dv = atanf(pw / (ph + EPSf)) - atanf(lw2 / (lh2 + EPSf));
    float vv = 0.40528473456935109f * dv * dv;     // 4/pi^2
    float alp = vv / (1.f - iou + vv + EPSf);
    float ciou = iou - rho2 / c2 - alp * vv;
    float scale = 2.f - lw2 * lh2 / img2;
    float aiou_v = scale * (1.f - ciou) * mix;

    float cx = sP[w][a * 85 + 4];                  // conf logit
    float spc = softplusf(cx);
    float sg  = 1.f / (1.f + __expf(-cx));
    float f1  = (spc - cx) * (1.f - sg) * (1.f - sg) * mix;  // focal(x,1)*mix
    float f0  = spc * sg * sg * mix;                          // focal(x,0)*mix

    // ---- class-BCE row sums via WMMA with ones-B, 2 interleaved tiles ----
    V16H ones;
    #pragma unroll
    for (int e = 0; e < 16; ++e) ones.h[e] = (_Float16)1.0f;

    const int row = lane & 15;
    const int ac0 = min(row,      nstage - 1);
    const int ac1 = min(16 + row, nstage - 1);
    const float* PX0 = &sP[w][ac0 * 85 + 5];
    const float* TX0 = &sL[w][ac0 * 86 + 6];
    const float* PX1 = &sP[w][ac1 * 85 + 5];
    const float* TX1 = &sL[w][ac1 * 86 + 6];

    V8F acc0, acc1;
    #pragma unroll
    for (int r = 0; r < 8; ++r) { acc0.f[r] = 0.f; acc1.f[r] = 0.f; }

    for (int c = 0; c < 3; ++c) {                  // 3 x K=32 chunks (80 + 16 pad)
        const int base = c * 32 + (lane & 16);     // lanes l, l+16 split the 32 k
        V16H a0, a1;
        #pragma unroll
        for (int e = 0; e < 16; ++e) {
            int ci = base + e;
            int cc = min(ci, NC - 1);              // branchless pad (clamped load)
            float x0v = PX0[cc], t0v = TX0[cc];
            float x1v = PX1[cc], t1v = TX1[cc];
            float v0e = softplusf(x0v) - x0v * t0v;
            float v1e = softplusf(x1v) - x1v * t1v;
            a0.h[e] = (_Float16)((ci < NC) ? v0e : 0.f);
            a1.h[e] = (_Float16)((ci < NC) ? v1e : 0.f);
        }
        acc0.v = __builtin_amdgcn_wmma_f32_16x16x32_f16(
            false, a0.v, false, ones.v, (short)0, acc0.v, false, false);
        acc1.v = __builtin_amdgcn_wmma_f32_16x16x32_f16(
            false, a1.v, false, ones.v, (short)0, acc1.v, false, false);
    }
    // D[m][n] = rowsum_m (replicated over n). lane0 holds rows 0-7,
    // lane16 holds rows 8-15 (one per VGPR) -> exchange through LDS.
    if (lane == 0) {
        #pragma unroll
        for (int r = 0; r < 8; ++r) { sX[w][r] = acc0.f[r]; sX[w][16 + r] = acc1.f[r]; }
    }
    if (lane == 16) {
        #pragma unroll
        for (int r = 0; r < 8; ++r) { sX[w][8 + r] = acc0.f[r]; sX[w][24 + r] = acc1.f[r]; }
    }
    asm volatile("s_wait_dscnt 0" ::: "memory");   // wave-local LDS RAW
    float clsm = sX[w][lane] * mix;

    if (lane < nstage) {
        long long o = n0 + lane;
        o_loss[o] = aiou_v + 0.5f * f1 + clsm;     // selection loss (paa anchor loss)
        o_aiou[o] = aiou_v;
        o_cls[o]  = clsm;
        o_f0[o]   = f0;
        o_f1[o]   = f1;
    }
}

// ============================================================
// K2: append each anchor's loss into its (b, j, layer) bucket.
// ============================================================
__global__ void k_bucket(const float* __restrict__ loss, const int* __restrict__ index,
                         int* __restrict__ cnt, float* __restrict__ vals,
                         int N, int layer, int totalBN)
{
    int t = blockIdx.x * blockDim.x + threadIdx.x;
    if (t >= totalBN) return;
    int j = index[t];
    if ((unsigned)j < (unsigned)Jg) {
        int b  = t / N;
        int bk = (b * Jg + j) * 3 + layer;
        int pos = atomicAdd(&cnt[bk], 1);
        if (pos < CAP) vals[bk * CAP + pos] = loss[t];
    }
}

// ============================================================
// K3: wave per bucket -> 15 smallest (ascending) + k_thresh.
// ============================================================
__global__ __launch_bounds__(128) void k_select(const int* __restrict__ cnt,
                                                const float* __restrict__ vals,
                                                float* __restrict__ cand,
                                                float* __restrict__ kthr)
{
    const int w = threadIdx.x >> 5, lane = threadIdx.x & 31;
    const int bk = blockIdx.x * 4 + w;
    if (bk >= Bb * Jg * 3) return;
    const float INF = __builtin_inff();
    const int c = min(cnt[bk], CAP);
    const float* v = vals + bk * CAP;
    float s0 = (lane      < c) ? v[lane]      : INF;
    float s1 = (lane + 32 < c) ? v[lane + 32] : INF;
    float s2 = (lane + 64 < c) ? v[lane + 64] : INF;
    const int bj = bk / 3, layer = bk % 3;
    float* cd = cand + bj * 45 + layer * KSEL;
    float kt = 0.f;
    for (int r = 0; r < KSEL; ++r) {
        float lm = fminf(s0, fminf(s1, s2));
        float bv = lm; int bl = lane;
        #pragma unroll
        for (int o = 16; o > 0; o >>= 1) {
            float ov = __shfl_xor(bv, o, 32);
            int   ol = __shfl_xor(bl, o, 32);
            if (ov < bv || (ov == bv && ol < bl)) { bv = ov; bl = ol; }
        }
        if (lane == 0) cd[r] = bv;                 // INF when exhausted
        if (bv < INF) kt = bv;                     // max of extracted (ascending)
        if (lane == bl && lm < INF) {
            if      (s0 == lm) s0 = INF;
            else if (s1 == lm) s1 = INF;
            else               s2 = INF;
        }
    }
    if (lane == 0) kthr[bk] = (c > 0) ? kt : 0.f;
}

// ============================================================
// K4: wave per (b,j): 2-component GMM, 30 EM iterations on the
// 45 candidates (2 slots per lane), then reference's threshold rule.
// ============================================================
__global__ __launch_bounds__(128) void k_gmm(const float* __restrict__ cand,
                                             float* __restrict__ thr)
{
    const int w = threadIdx.x >> 5, lane = threadIdx.x & 31;
    const int id = blockIdx.x * 4 + w;
    if (id >= Bb * Jg) return;
    const float INF = __builtin_inff();
    const float* cd = cand + id * 45;
    float x0 = cd[lane];
    float x1 = (lane + 32 < 45) ? cd[lane + 32] : INF;
    const bool v0 = x0 < INF, v1 = x1 < INF;
    const float nf = wsum((v0 ? 1.f : 0.f) + (v1 ? 1.f : 0.f));
    float mn = wmin(fminf(x0, x1));
    float mx = wmax(fmaxf(v0 ? x0 : -INF, v1 ? x1 : -INF));
    if (nf < 0.5f) { mn = 0.f; mx = 1.f; }
    const float xs0 = v0 ? x0 : 0.f, xs1 = v1 ? x1 : 0.f;

    float mu0 = mn, mu1 = mx, va0 = 1.f, va1 = 1.f, w0 = .5f, w1 = .5f;
    const float L2PI = 1.8378770664093453f;        // log(2*pi)
    for (int it = 0; it < EMIT; ++it) {
        float lw0 = __logf(w0), lw1 = __logf(w1);
        float lv0 = __logf(va0), lv1 = __logf(va1);
        float r00 = 0.f, r10 = 0.f, r01 = 0.f, r11 = 0.f;
        if (v0) {
            float d0 = x0 - mu0, d1 = x0 - mu1;
            float lp0 = -0.5f * (d0 * d0 / va0 + L2PI + lv0) + lw0;
            float lp1 = -0.5f * (d1 * d1 / va1 + L2PI + lv1) + lw1;
            float r = 1.f / (1.f + __expf(lp1 - lp0));
            r00 = r; r10 = 1.f - r;
        }
        if (v1) {
            float d0 = x1 - mu0, d1 = x1 - mu1;
            float lp0 = -0.5f * (d0 * d0 / va0 + L2PI + lv0) + lw0;
            float lp1 = -0.5f * (d1 * d1 / va1 + L2PI + lv1) + lw1;
            float r = 1.f / (1.f + __expf(lp1 - lp0));
            r01 = r; r11 = 1.f - r;
        }
        float nk0 = wsum(r00 + r01) + 1e-10f;
        float nk1 = wsum(r10 + r11) + 1e-10f;
        float s10 = wsum(r00 * xs0 + r01 * xs1);
        float s11 = wsum(r10 * xs0 + r11 * xs1);
        mu0 = s10 / nk0; mu1 = s11 / nk1;
        float q0 = r00 * (xs0 - mu0) * (xs0 - mu0) + r01 * (xs1 - mu0) * (xs1 - mu0);
        float q1 = r10 * (xs0 - mu1) * (xs0 - mu1) + r11 * (xs1 - mu1) * (xs1 - mu1);
        va0 = wsum(q0) / nk0 + 1e-6f;
        va1 = wsum(q1) / nk1 + 1e-6f;
        float dn = fmaxf(nf, 1.f);
        w0 = nk0 / dn; w1 = nk1 / dn;
    }
    // final scoring: fg = argmax-comp==0; pick max score, tie -> smaller x
    float bs = -INF, bx = 0.f;
    float lw0 = __logf(w0), lw1 = __logf(w1);
    float lv0 = __logf(va0), lv1 = __logf(va1);
    #pragma unroll
    for (int s = 0; s < 2; ++s) {
        bool  vs = s ? v1 : v0;
        float xv = s ? x1 : x0;
        if (vs) {
            float d0 = xv - mu0, d1 = xv - mu1;
            float lp0 = -0.5f * (d0 * d0 / va0 + L2PI + lv0) + lw0;
            float lp1 = -0.5f * (d1 * d1 / va1 + L2PI + lv1) + lw1;
            if (lp0 >= lp1) {                      // comp 0 (argmax, first occ.)
                float m = fmaxf(lp0, lp1);
                float sc = m + __logf(__expf(lp0 - m) + __expf(lp1 - m));
                if (sc > bs || (sc == bs && xv < bx)) { bs = sc; bx = xv; }
            }
        }
    }
    #pragma unroll
    for (int o = 16; o > 0; o >>= 1) {
        float ov = __shfl_xor(bs, o, 32);
        float ox = __shfl_xor(bx, o, 32);
        if (ov > bs || (ov == bs && ox < bx)) { bs = ov; bx = ox; }
    }
    float res;
    if (nf > 1.5f)      res = (bs > -INF) ? bx : mn;
    else if (nf > 0.5f) res = mn;
    else                res = 0.f;
    if (lane == 0) thr[id] = res;
}

// ============================================================
// K5: masked per-layer reduction -> per-block partials (fixed tree).
// ============================================================
__global__ __launch_bounds__(256) void k_mask_reduce(
    const float* __restrict__ loss, const float* __restrict__ aiou,
    const float* __restrict__ clsm, const float* __restrict__ f0o,
    const float* __restrict__ f1o,  const int* __restrict__ index,
    const float* __restrict__ thr,  const float* __restrict__ kthr,
    float* __restrict__ partials, int N, int layer, int totalBN, int pbase)
{
    __shared__ float r0[256], r1[256], r2[256];
    const int tid = threadIdx.x;
    const int t = blockIdx.x * 256 + tid;
    float li = 0.f, lc = 0.f, lf = 0.f;
    if (t < totalBN) {
        float lv = loss[t];
        int   j  = index[t];
        bool  m  = (lv < 0.f);
        if ((unsigned)j < (unsigned)Jg) {
            int b = t / N;
            float th = fminf(thr[b * Jg + j], kthr[(b * Jg + j) * 3 + layer]);
            m |= (lv <= th);
        }
        li = m ? aiou[t] : 0.f;
        lc = m ? clsm[t] : 0.f;
        lf = m ? f1o[t]  : f0o[t];                 // focal(x, mask)*mix
    }
    r0[tid] = li; r1[tid] = lc; r2[tid] = lf;
    __syncthreads();
    for (int s = 128; s > 0; s >>= 1) {
        if (tid < s) { r0[tid] += r0[tid + s]; r1[tid] += r1[tid + s]; r2[tid] += r2[tid + s]; }
        __syncthreads();
    }
    if (tid == 0) {
        int o = (pbase + blockIdx.x) * 3;
        partials[o + 0] = r0[0];   // iou
        partials[o + 1] = r1[0];   // cls
        partials[o + 2] = r2[0];   // conf
    }
}

// K6: deterministic fixed-order final sum -> 4 output scalars.
__global__ void k_final(const float* __restrict__ partials, int nblk, float* __restrict__ out)
{
    if (threadIdx.x == 0 && blockIdx.x == 0) {
        float siou = 0.f, scls = 0.f, sconf = 0.f;
        for (int i = 0; i < nblk; ++i) {
            siou  += partials[i * 3 + 0];
            scls  += partials[i * 3 + 1];
            sconf += partials[i * 3 + 2];
        }
        float liou  = siou  / (float)Bb;
        float lcls  = scls  / (float)Bb;
        float lconf = sconf / (float)Bb;
        out[0] = liou + lconf + lcls;
        out[1] = liou;
        out[2] = lconf;
        out[3] = lcls;
    }
}

// ============================================================
extern "C" void kernel_launch(void* const* d_in, const int* in_sizes, int n_in,
                              void* d_out, int out_size, void* d_ws, size_t ws_size,
                              hipStream_t stream)
{
    // setup_inputs dict order: p_s, p_d_s, label_s, index_s, (m...), (l...), bboxes
    const float*  p_[3]   = {(const float*)d_in[0], (const float*)d_in[4], (const float*)d_in[8]};
    const float4* pd_[3]  = {(const float4*)d_in[1], (const float4*)d_in[5], (const float4*)d_in[9]};
    const float*  lab_[3] = {(const float*)d_in[2], (const float*)d_in[6], (const float*)d_in[10]};
    const int*    idx_[3] = {(const int*)d_in[3], (const int*)d_in[7], (const int*)d_in[11]};
    const int   Nl[3]   = {52 * 52 * 3, 26 * 26 * 3, 13 * 13 * 3}; // 8112, 2028, 507
    const float im2     = 416.f * 416.f;                            // g*stride = 416 for all
    const int   offl[3] = {0, 8112 * Bb, (8112 + 2028) * Bb};
    const size_t NT = (size_t)(8112 + 2028 + 507) * Bb;             // 170,352 anchors

    float* ws     = (float*)d_ws;
    float* w_loss = ws;
    float* w_aiou = ws + NT;
    float* w_cls  = ws + 2 * NT;
    float* w_f0   = ws + 3 * NT;
    float* w_f1   = ws + 4 * NT;
    int*   w_cnt  = (int*)(ws + 5 * NT);                 // 2880
    float* w_vals = (float*)(w_cnt + Bb * Jg * 3);       // 2880*96
    float* w_cand = w_vals + Bb * Jg * 3 * CAP;          // 16*60*45
    float* w_kthr = w_cand + Bb * Jg * 45;               // 2880
    float* w_thr  = w_kthr + Bb * Jg * 3;                // 960
    float* w_part = w_thr + Bb * Jg;                     // <= 666*3

    hipMemsetAsync(w_cnt, 0, Bb * Jg * 3 * sizeof(int), stream);

    for (int L = 0; L < 3; ++L) {
        int tot    = Nl[L] * Bb;
        int waves  = (tot + 31) / 32;
        int blocks = (waves + 1) / 2;                    // 64-thread blocks, 2 waves
        k_anchor<<<blocks, 64, 0, stream>>>(p_[L], pd_[L], lab_[L],
            w_loss + offl[L], w_aiou + offl[L], w_cls + offl[L],
            w_f0 + offl[L], w_f1 + offl[L], tot, im2);
    }
    for (int L = 0; L < 3; ++L) {
        int tot = Nl[L] * Bb;
        k_bucket<<<(tot + 255) / 256, 256, 0, stream>>>(
            w_loss + offl[L], idx_[L], w_cnt, w_vals, Nl[L], L, tot);
    }
    k_select<<<(Bb * Jg * 3 + 3) / 4, 128, 0, stream>>>(w_cnt, w_vals, w_cand, w_kthr);
    k_gmm<<<(Bb * Jg + 3) / 4, 128, 0, stream>>>(w_cand, w_thr);

    int pbase = 0;
    for (int L = 0; L < 3; ++L) {
        int tot    = Nl[L] * Bb;
        int blocks = (tot + 255) / 256;
        k_mask_reduce<<<blocks, 256, 0, stream>>>(
            w_loss + offl[L], w_aiou + offl[L], w_cls + offl[L],
            w_f0 + offl[L], w_f1 + offl[L], idx_[L],
            w_thr, w_kthr, w_part, Nl[L], L, tot, pbase);
        pbase += blocks;
    }
    k_final<<<1, 32, 0, stream>>>(w_part, pbase, (float*)d_out);
}
